// Attention_45337674777236
// MI455X (gfx1250) — matrix operand order repaired and checked by reading, not compiled, not run on hardware
//
#include <hip/hip_runtime.h>
#include <hip/hip_bf16.h>

typedef __attribute__((ext_vector_type(16))) __bf16 v16bf;
typedef __attribute__((ext_vector_type(8)))  float  v8f;

#define B_      4
#define C_      128
#define L_      4096      // H*W
#define HEADS_  4
#define CH_     32        // C / HEADS
#define GROUPS_ 32
#define CPG_    4         // C / GROUPS
#define EPS_    1e-5f
#define SCALE_  0.4204482076268573f   // 1/sqrt(sqrt(32))

// ---------------- WMMA helpers (16x16x32 bf16 -> f32) ----------------

__device__ inline v8f wmma_bf16(v16bf a, v16bf b, v8f c) {
  return __builtin_amdgcn_wmma_f32_16x16x32_bf16(false, a, false, b, (short)0, c,
                                                 false, false);
}

__device__ inline v8f zero8() {
  v8f z;
#pragma unroll
  for (int i = 0; i < 8; ++i) z[i] = 0.f;
  return z;
}

union Frag16 {
  uint4 q[2];
  v16bf v;
};

// A fragment, 16(M) x 32(K), row-major storage: A[m][k] = base[(mBase+m)*ld + kBase+k]
// Per lane this is two contiguous 16B runs: k in [khalf, khalf+8) and [khalf+16, khalf+24).
__device__ inline v16bf load_frag_a_rm(const __bf16* base, int ld, int mBase, int kBase) {
  int lane = threadIdx.x & 31;
  const __bf16* p = base + (size_t)(mBase + (lane & 15)) * ld + kBase + ((lane >> 4) << 3);
  Frag16 f;
  f.q[0] = *(const uint4*)(p);
  f.q[1] = *(const uint4*)(p + 16);
  return f.v;
}

// B fragment, 32(K) x 16(N), N-major storage: B[k][n] = base[(nBase+n)*ld + kBase+k]
// Per lane this is two contiguous 16B runs: k in [khalf, khalf+8) and [khalf+16, khalf+24).
__device__ inline v16bf load_frag_b_nm(const __bf16* base, int ld, int kBase, int nBase) {
  int lane = threadIdx.x & 31;
  const __bf16* p = base + (size_t)(nBase + (lane & 15)) * ld + kBase + ((lane >> 4) << 3);
  Frag16 f;
  f.q[0] = *(const uint4*)(p);
  f.q[1] = *(const uint4*)(p + 16);
  return f.v;
}

// ---------------- Kernel 1: GroupNorm statistics ----------------
__global__ void gn_stats_kernel(const float* __restrict__ x,
                                float* __restrict__ mean_o,
                                float* __restrict__ rstd_o) {
  const int bg = blockIdx.x;
  const float* p = x + (size_t)bg * (CPG_ * L_);
  float s = 0.f, sq = 0.f;
  for (int i = threadIdx.x; i < CPG_ * L_; i += blockDim.x) {
    float v = p[i];
    s += v;
    sq += v * v;
  }
  __shared__ float ls[256], lq[256];
  ls[threadIdx.x] = s;
  lq[threadIdx.x] = sq;
  __syncthreads();
  for (int o = 128; o > 0; o >>= 1) {
    if ((int)threadIdx.x < o) {
      ls[threadIdx.x] += ls[threadIdx.x + o];
      lq[threadIdx.x] += lq[threadIdx.x + o];
    }
    __syncthreads();
  }
  if (threadIdx.x == 0) {
    const float inv = 1.0f / (float)(CPG_ * L_);
    float m = ls[0] * inv;
    float var = lq[0] * inv - m * m;
    mean_o[bg] = m;
    rstd_o[bg] = rsqrtf(var + EPS_);
  }
}

// ---------------- Kernel 2: GN apply + bf16 cast + transpose -> hn_t[b][l][c] ----
// 32x32 tile transpose through LDS; both global sides coalesced.
__global__ __launch_bounds__(256) void gn_apply_kernel(const float* __restrict__ x,
                                                       const float* __restrict__ gamma,
                                                       const float* __restrict__ beta,
                                                       const float* __restrict__ mean,
                                                       const float* __restrict__ rstd,
                                                       __bf16* __restrict__ hnt) {
  const int b = blockIdx.z;
  const int c0 = blockIdx.y * 32;
  const int l0 = blockIdx.x * 32;
  const int tx = threadIdx.x;  // 0..31
  const int ty = threadIdx.y;  // 0..7
  __shared__ __bf16 tile[32][33];
#pragma unroll
  for (int i = 0; i < 4; ++i) {
    int c = c0 + ty + i * 8;
    int g = c >> 2;
    float m = mean[b * GROUPS_ + g];
    float rs = rstd[b * GROUPS_ + g];
    float v = (x[((size_t)b * C_ + c) * L_ + l0 + tx] - m) * rs * gamma[c] + beta[c];
    tile[ty + i * 8][tx] = (__bf16)v;
  }
  __syncthreads();
#pragma unroll
  for (int i = 0; i < 4; ++i) {
    int l = l0 + ty + i * 8;
    hnt[((size_t)b * L_ + l) * C_ + c0 + tx] = tile[tx][ty + i * 8];
  }
}

// ---------------- Kernel 3: weight casts ----------------
__global__ void wconv_kernel(const float* __restrict__ wq, const float* __restrict__ wp,
                             __bf16* __restrict__ wqb, __bf16* __restrict__ wpb) {
  int idx = blockIdx.x * blockDim.x + threadIdx.x;
  if (idx < 3 * C_ * C_) {
    wqb[idx] = (__bf16)wq[idx];
  } else if (idx < 4 * C_ * C_) {
    wpb[idx - 3 * C_ * C_] = (__bf16)wp[idx - 3 * C_ * C_];
  }
}

// ---------------- Kernel 4: QKV GEMM, transposed product ----------------
// D[l][o] = sum_c hn_t[b][l][c] * wq[o][c]   (M=4096 rows l, N=384 cols o, K=128)
// A = hn_t (row-major), B = wq (N-major). Fuses bias + attention scale.
// Outputs: q_t[bh][l][c], k_t[bh][l][c] (contiguous writes), v_t[bh][c][l].
__global__ __launch_bounds__(256) void qkv_gemm_kernel(const __bf16* __restrict__ hnt,
                                                       const __bf16* __restrict__ wq,
                                                       const float* __restrict__ bias,
                                                       __bf16* __restrict__ qt,
                                                       __bf16* __restrict__ kt,
                                                       __bf16* __restrict__ vt) {
  const int b = blockIdx.z;
  const int blockM = blockIdx.x * 128;  // l
  const int blockN = blockIdx.y * 128;  // o
  const int wave = threadIdx.x >> 5;
  const int mSub = wave >> 1;  // 0..3
  const int nSub = wave & 1;   // 0..1
  const int lane = threadIdx.x & 31;
  const __bf16* Ap = hnt + (size_t)b * L_ * C_;

  v8f acc[2][4];
#pragma unroll
  for (int mt = 0; mt < 2; ++mt)
#pragma unroll
    for (int nt = 0; nt < 4; ++nt) acc[mt][nt] = zero8();

  for (int kk = 0; kk < C_; kk += 32) {
    v16bf aF[2], bF[4];
#pragma unroll
    for (int mt = 0; mt < 2; ++mt)
      aF[mt] = load_frag_a_rm(Ap, C_, blockM + mSub * 32 + mt * 16, kk);
#pragma unroll
    for (int nt = 0; nt < 4; ++nt)
      bF[nt] = load_frag_b_nm(wq, C_, kk, blockN + nSub * 64 + nt * 16);
#pragma unroll
    for (int mt = 0; mt < 2; ++mt)
#pragma unroll
      for (int nt = 0; nt < 4; ++nt) acc[mt][nt] = wmma_bf16(aF[mt], bF[nt], acc[mt][nt]);
  }

#pragma unroll
  for (int mt = 0; mt < 2; ++mt)
#pragma unroll
    for (int nt = 0; nt < 4; ++nt)
#pragma unroll
      for (int r = 0; r < 8; ++r) {
        int l = blockM + mSub * 32 + mt * 16 + r + ((lane >> 4) << 3);
        int o = blockN + nSub * 64 + nt * 16 + (lane & 15);
        float vv = acc[mt][nt][r] + bias[o];
        if (o < 2 * C_) vv *= SCALE_;
        __bf16 bv = (__bf16)vv;
        if (o < C_) {
          int bh = b * HEADS_ + (o >> 5);
          qt[((size_t)bh * L_ + l) * CH_ + (o & 31)] = bv;
        } else if (o < 2 * C_) {
          int oc = o - C_;
          int bh = b * HEADS_ + (oc >> 5);
          kt[((size_t)bh * L_ + l) * CH_ + (oc & 31)] = bv;
        } else {
          int oc = o - 2 * C_;
          int bh = b * HEADS_ + (oc >> 5);
          vt[((size_t)bh * CH_ + (oc & 31)) * L_ + l] = bv;
        }
      }
}

// ---------------- Kernel 5: flash attention ----------------
// One wave per block owns 64 query rows (4 x 16-row tiles) of one (b,head).
__global__ __launch_bounds__(32) void attn_kernel(const __bf16* __restrict__ qt,
                                                  const __bf16* __restrict__ kt,
                                                  const __bf16* __restrict__ vt,
                                                  __bf16* __restrict__ at) {
  const int bh = blockIdx.y;
  const int b = bh >> 2;
  const int h = bh & 3;
  const int rowBase = blockIdx.x * 64;
  const __bf16* qp = qt + (size_t)bh * L_ * CH_;  // [l][c] row-major A
  const __bf16* kp = kt + (size_t)bh * L_ * CH_;  // [s][c] = N-major B (n=s,k=c)
  const __bf16* vp = vt + (size_t)bh * CH_ * L_;  // [c][s] = N-major B (n=c,k=s)

  __shared__ __align__(16) float  s_buf[16][32];
  __shared__ __align__(16) __bf16 p_buf[16][32];
  __shared__ float bcast[16];

  const int lane = threadIdx.x;
  const int row = lane & 15;
  const int half = lane >> 4;

  v16bf qF[4];
#pragma unroll
  for (int t = 0; t < 4; ++t) qF[t] = load_frag_a_rm(qp, CH_, rowBase + t * 16, 0);

  v8f o[4][2];
  float m_run[4], l_run[4];
#pragma unroll
  for (int t = 0; t < 4; ++t) {
    o[t][0] = zero8();
    o[t][1] = zero8();
    m_run[t] = -1e30f;
    l_run[t] = 0.f;
  }

  for (int s0 = 0; s0 < L_; s0 += 32) {
    v16bf kF0 = load_frag_b_nm(kp, CH_, 0, s0);
    v16bf kF1 = load_frag_b_nm(kp, CH_, 0, s0 + 16);
    v16bf vF0 = load_frag_b_nm(vp, L_, s0, 0);
    v16bf vF1 = load_frag_b_nm(vp, L_, s0, 16);

#pragma unroll
    for (int t = 0; t < 4; ++t) {
      v8f S0 = wmma_bf16(qF[t], kF0, zero8());
      v8f S1 = wmma_bf16(qF[t], kF1, zero8());
#pragma unroll
      for (int r = 0; r < 8; ++r) {
        int m = r + (half << 3);
        s_buf[m][row]      = S0[r];
        s_buf[m][row + 16] = S1[r];
      }
      // softmax: lane (row,half) handles 16 of 32 columns; read back as float4s
      const int cb = half << 4;
      float vals[16];
      {
        const float4* sr = (const float4*)(&s_buf[row][cb]);
#pragma unroll
        for (int q4 = 0; q4 < 4; ++q4) {
          float4 f = sr[q4];
          vals[4 * q4 + 0] = f.x;
          vals[4 * q4 + 1] = f.y;
          vals[4 * q4 + 2] = f.z;
          vals[4 * q4 + 3] = f.w;
        }
      }
      float mx = -1e30f;
#pragma unroll
      for (int j = 0; j < 16; ++j) mx = fmaxf(mx, vals[j]);
      mx = fmaxf(mx, __shfl_xor(mx, 16, 32));
      float m_new = fmaxf(m_run[t], mx);
      float alpha = __expf(m_run[t] - m_new);
      float ls = 0.f;
      unsigned* pw = (unsigned*)(&p_buf[row][cb]);
#pragma unroll
      for (int jj = 0; jj < 8; ++jj) {
        float p0 = __expf(vals[2 * jj + 0] - m_new);
        float p1 = __expf(vals[2 * jj + 1] - m_new);
        ls += p0 + p1;
        unsigned short u0 = __builtin_bit_cast(unsigned short, (__bf16)p0);
        unsigned short u1 = __builtin_bit_cast(unsigned short, (__bf16)p1);
        pw[jj] = ((unsigned)u1 << 16) | (unsigned)u0;
      }
      ls += __shfl_xor(ls, 16, 32);
      l_run[t] = l_run[t] * alpha + ls;
      m_run[t] = m_new;
      if (half == 0) bcast[row] = alpha;
#pragma unroll
      for (int r = 0; r < 8; ++r) {
        float sc = bcast[r + (half << 3)];
        o[t][0][r] *= sc;
        o[t][1][r] *= sc;
      }
      v16bf pF = load_frag_a_rm(&p_buf[0][0], 32, 0, 0);
      o[t][0] = wmma_bf16(pF, vF0, o[t][0]);
      o[t][1] = wmma_bf16(pF, vF1, o[t][1]);
    }
  }

  // finalize: divide by row sums; write a_t[b][l][c] (contiguous 32B runs)
#pragma unroll
  for (int t = 0; t < 4; ++t) {
    if (half == 0) bcast[row] = 1.0f / l_run[t];
#pragma unroll
    for (int ch2 = 0; ch2 < 2; ++ch2)
#pragma unroll
      for (int r = 0; r < 8; ++r) {
        int trow = rowBase + t * 16 + r + (half << 3);
        int c = (ch2 << 4) + row;
        float val = o[t][ch2][r] * bcast[r + (half << 3)];
        at[((size_t)b * L_ + trow) * C_ + h * CH_ + c] = (__bf16)val;
      }
  }
}

// ---------------- Kernel 6: proj GEMM + bias + residual ----------------
// out[b][o][l] = x + b_proj[o] + sum_c wp[o][c] * a_t[b][l][c]
// A = wp (row-major), B = a_t (N-major, n=l).
__global__ __launch_bounds__(256) void proj_kernel(const __bf16* __restrict__ wp,
                                                   const __bf16* __restrict__ at,
                                                   const float* __restrict__ bias,
                                                   const float* __restrict__ x,
                                                   float* __restrict__ out) {
  const int b = blockIdx.y;
  const int blockN = blockIdx.x * 128;  // l
  const int wave = threadIdx.x >> 5;
  const int mSub = wave >> 1;
  const int nSub = wave & 1;
  const int lane = threadIdx.x & 31;
  const __bf16* Bp = at + (size_t)b * L_ * C_;

  v8f acc[2][4];
#pragma unroll
  for (int mt = 0; mt < 2; ++mt)
#pragma unroll
    for (int nt = 0; nt < 4; ++nt) acc[mt][nt] = zero8();

  for (int kk = 0; kk < C_; kk += 32) {
    v16bf aF[2], bF[4];
#pragma unroll
    for (int mt = 0; mt < 2; ++mt)
      aF[mt] = load_frag_a_rm(wp, C_, mSub * 32 + mt * 16, kk);
#pragma unroll
    for (int nt = 0; nt < 4; ++nt)
      bF[nt] = load_frag_b_nm(Bp, C_, kk, blockN + nSub * 64 + nt * 16);
#pragma unroll
    for (int mt = 0; mt < 2; ++mt)
#pragma unroll
      for (int nt = 0; nt < 4; ++nt) acc[mt][nt] = wmma_bf16(aF[mt], bF[nt], acc[mt][nt]);
  }

#pragma unroll
  for (int mt = 0; mt < 2; ++mt)
#pragma unroll
    for (int nt = 0; nt < 4; ++nt)
#pragma unroll
      for (int r = 0; r < 8; ++r) {
        int oc = mSub * 32 + mt * 16 + r + ((lane >> 4) << 3);
        int n = blockN + nSub * 64 + nt * 16 + (lane & 15);
        size_t gi = ((size_t)b * C_ + oc) * L_ + n;
        out[gi] = x[gi] + acc[mt][nt][r] + bias[oc];
      }
}

// ---------------- launch ----------------
extern "C" void kernel_launch(void* const* d_in, const int* in_sizes, int n_in,
                              void* d_out, int out_size, void* d_ws, size_t ws_size,
                              hipStream_t stream) {
  const float* x      = (const float*)d_in[0];
  const float* gamma  = (const float*)d_in[1];
  const float* beta   = (const float*)d_in[2];
  const float* w_qkv  = (const float*)d_in[3];
  const float* b_qkv  = (const float*)d_in[4];
  const float* w_proj = (const float*)d_in[5];
  const float* b_proj = (const float*)d_in[6];
  float* out = (float*)d_out;

  char* ws = (char*)d_ws;
  float*  mean = (float*)(ws + 0);            // 128 f32
  float*  rstd = (float*)(ws + 512);          // 128 f32
  __bf16* wqb  = (__bf16*)(ws + 1024);        // 384*128 bf16
  __bf16* wpb  = (__bf16*)(ws + 99328);       // 128*128 bf16
  __bf16* hnt  = (__bf16*)(ws + 132096);      // [b][l][c]   4 MiB
  __bf16* qt   = (__bf16*)(ws + 4326400);     // [bh][l][ch] 4 MiB
  __bf16* kt   = (__bf16*)(ws + 8520704);     // [bh][l][ch] 4 MiB
  __bf16* vt   = (__bf16*)(ws + 12715008);    // [bh][ch][l] 4 MiB
  __bf16* at   = (__bf16*)(ws + 16909312);    // [b][l][c]   4 MiB
  (void)ws_size; (void)in_sizes; (void)n_in; (void)out_size;

  gn_stats_kernel<<<B_ * GROUPS_, 256, 0, stream>>>(x, mean, rstd);
  gn_apply_kernel<<<dim3(L_ / 32, C_ / 32, B_), dim3(32, 8), 0, stream>>>(
      x, gamma, beta, mean, rstd, hnt);
  wconv_kernel<<<(4 * C_ * C_) / 256, 256, 0, stream>>>(w_qkv, w_proj, wqb, wpb);
  qkv_gemm_kernel<<<dim3(32, 3, B_), 256, 0, stream>>>(hnt, wqb, b_qkv, qt, kt, vt);
  attn_kernel<<<dim3(L_ / 64, B_ * HEADS_), 32, 0, stream>>>(qt, kt, vt, at);
  proj_kernel<<<dim3(32, B_), 256, 0, stream>>>(wpb, at, b_proj, x, out);
}
